// edge_exists_predictor_35734127903067
// MI455X (gfx1250) — compile-verified, matrix operand order checked by reading, not run on hardware
//
#include <hip/hip_runtime.h>
#include <hip/hip_bf16.h>

typedef __attribute__((ext_vector_type(16))) __bf16 v16bf;
typedef __attribute__((ext_vector_type(8)))  __bf16 v8bf;
typedef __attribute__((ext_vector_type(4)))  __bf16 v4bf;
typedef __attribute__((ext_vector_type(8)))  float  v8f;

#define LD 64   // padded leading dimension of all node-feature buffers

// ---------------- degree / norm ----------------
__global__ void k_fill_ones(float* __restrict__ p, int n) {
  int i = blockIdx.x * blockDim.x + threadIdx.x;
  if (i < n) p[i] = 1.0f;
}

__global__ void k_count_deg(const int* __restrict__ dst, float* __restrict__ deg, long long E) {
  long long i = (long long)blockIdx.x * blockDim.x + threadIdx.x;
  if (i < E) atomicAdd(&deg[dst[i]], 1.0f);
}

__global__ void k_rsqrt_inplace(float* __restrict__ p, int n) {
  int i = blockIdx.x * blockDim.x + threadIdx.x;
  if (i < n) p[i] = rsqrtf(p[i]);
}

// ---------------- fused WMMA GEMM + self-loop/bias epilogue ----------------
// H[N,LD]  = X[nrows,kdim] @ W[kdim,wcols]        (raw conv features)
// A[N,LD]  = H * dinv[row]^2 + bias[col]          (agg init: self-loop + bias)
// Block: 128 threads = 4 waves; 64 rows x 64 (zero-padded) cols per block.
// B stored TRANSPOSED in LDS so every fragment half is 8 contiguous bf16
// (one ds_load_b128). Columns >= wcols compute exact zeros (B zero-padded),
// so stores are unguarded full-width float4 through an LDS C-tile.
__global__ void __launch_bounds__(128)
k_gemm_fused(const float* __restrict__ X, int ldx, int kdim,
             const float* __restrict__ W, int wcols,
             const float* __restrict__ dinv, const float* __restrict__ bias,
             float* __restrict__ H, float* __restrict__ A, int nrows)
{
  __shared__ __align__(16) __bf16 sA[64][64];    // A tile, row-major
  __shared__ __align__(16) __bf16 sBt[64][64];   // B tile, TRANSPOSED: sBt[n][k]
  __shared__ __align__(16) float  sC[64][64];    // C staging for coalesced stores
  __shared__ float sDinv[64];
  __shared__ float sBias[64];

  const int tid  = threadIdx.x;
  const int row0 = blockIdx.x * 64;

  // ---- stage A (vector loads, fp32->bf16), kdim is a multiple of 4
  for (int i = tid; i < 64 * 16; i += 128) {
    int r = i >> 4, c4 = (i & 15) << 2;
    int gr = row0 + r;
    float4 v = make_float4(0.f, 0.f, 0.f, 0.f);
    if (gr < nrows && c4 < kdim) v = *(const float4*)&X[(size_t)gr * ldx + c4];
    v4bf b = {(__bf16)v.x, (__bf16)v.y, (__bf16)v.z, (__bf16)v.w};
    *(v4bf*)&sA[r][c4] = b;
  }
  // ---- stage B transposed (coalesced global read, scalar LDS write)
  for (int i = tid; i < 64 * 64; i += 128) {
    int r = i >> 6, c = i & 63;
    float vb = 0.f;
    if (r < kdim && c < wcols) vb = W[r * wcols + c];
    sBt[c][r] = (__bf16)vb;
  }
  // ---- stage dinv / bias (zero-padded)
  if (tid < 64) {
    int gr = row0 + tid;
    sDinv[tid] = (gr < nrows) ? dinv[gr] : 0.f;
  } else {
    int c = tid - 64;
    sBias[c] = (c < wcols) ? bias[c] : 0.f;
  }
  __syncthreads();

  const int lane = tid & 31;
  const int wave = tid >> 5;
  const int half = lane >> 4;
  const int l16  = lane & 15;
  const int arow = wave * 16 + l16;

  // A fragments: per CDNA5 16-bit A 16x32 layout, each half of the fragment
  // is 8 contiguous K values -> two ds_load_b128 per k-step.
  v8bf a0l = *(const v8bf*)&sA[arow][half * 8];
  v8bf a0h = *(const v8bf*)&sA[arow][half * 8 + 16];
  v8bf a1l = *(const v8bf*)&sA[arow][half * 8 + 32];
  v8bf a1h = *(const v8bf*)&sA[arow][half * 8 + 48];
  v16bf a0 = __builtin_shufflevector(a0l, a0h, 0,1,2,3,4,5,6,7,8,9,10,11,12,13,14,15);
  v16bf a1 = __builtin_shufflevector(a1l, a1h, 0,1,2,3,4,5,6,7,8,9,10,11,12,13,14,15);

#pragma unroll
  for (int nt = 0; nt < 4; ++nt) {
    const int n0 = nt * 16;
    const __bf16* bp = &sBt[n0 + l16][0];
    v8bf b0l = *(const v8bf*)(bp + half * 8);
    v8bf b0h = *(const v8bf*)(bp + half * 8 + 16);
    v8bf b1l = *(const v8bf*)(bp + half * 8 + 32);
    v8bf b1h = *(const v8bf*)(bp + half * 8 + 48);
    v16bf b0 = __builtin_shufflevector(b0l, b0h, 0,1,2,3,4,5,6,7,8,9,10,11,12,13,14,15);
    v16bf b1 = __builtin_shufflevector(b1l, b1h, 0,1,2,3,4,5,6,7,8,9,10,11,12,13,14,15);

    v8f acc = {0.f, 0.f, 0.f, 0.f, 0.f, 0.f, 0.f, 0.f};
    acc = __builtin_amdgcn_wmma_f32_16x16x32_bf16(false, a0, false, b0,
                                                  (short)0, acc, false, false);
    acc = __builtin_amdgcn_wmma_f32_16x16x32_bf16(false, a1, false, b1,
                                                  (short)0, acc, false, false);
#pragma unroll
    for (int r = 0; r < 8; ++r)
      sC[wave * 16 + half * 8 + r][n0 + l16] = acc[r];
  }
  __syncthreads();

  // ---- fused epilogue: coalesced float4 stores of H and A
  for (int i = tid; i < 64 * 16; i += 128) {
    int r = i >> 4, c4 = (i & 15) << 2;
    int gr = row0 + r;
    if (gr >= nrows) continue;
    float4 v  = *(const float4*)&sC[r][c4];
    float4 bb = *(const float4*)&sBias[c4];
    float  w  = sDinv[r] * sDinv[r];
    float4 a;
    a.x = v.x * w + bb.x;
    a.y = v.y * w + bb.y;
    a.z = v.z * w + bb.z;
    a.w = v.w * w + bb.w;
    *(float4*)&H[(size_t)gr * LD + c4] = v;
    *(float4*)&A[(size_t)gr * LD + c4] = a;
  }
}

// ---------------- edge scatter: agg[dst] += h[src] * dinv[src]*dinv[dst] ----------------
// 16 lanes per edge; each lane gathers one float4 of the source row.
__global__ void k_edge_scatter(const int* __restrict__ src, const int* __restrict__ dst,
                               const float* __restrict__ dinv,
                               const float* __restrict__ h, float* __restrict__ agg,
                               int fdim, long long E)
{
  long long idx = (long long)blockIdx.x * blockDim.x + threadIdx.x;
  long long e = idx >> 4;
  if (e >= E) return;
  int f0 = (int)(idx & 15) << 2;
  if (f0 >= fdim) return;
  int s = src[e], d = dst[e];
  float w = dinv[s] * dinv[d];
  float4 v = *(const float4*)&h[(size_t)s * LD + f0];   // padding cols are zeros
  float* ad = agg + (size_t)d * LD + f0;
  atomicAdd(ad + 0, v.x * w);
  if (f0 + 1 < fdim) atomicAdd(ad + 1, v.y * w);
  if (f0 + 2 < fdim) atomicAdd(ad + 2, v.z * w);
  if (f0 + 3 < fdim) atomicAdd(ad + 3, v.w * w);
}

// ---------------- pooling ----------------
__global__ void k_zero(float* __restrict__ p, long long n) {
  long long i = (long long)blockIdx.x * blockDim.x + threadIdx.x;
  if (i < n) p[i] = 0.f;
}

__global__ void k_pool(const int* __restrict__ batch, const float* __restrict__ agg,
                       float* __restrict__ g, int fdim, int n)
{
  long long idx = (long long)blockIdx.x * blockDim.x + threadIdx.x;
  int node = (int)(idx >> 4);
  int f0   = ((int)(idx & 15)) << 2;
  if (node >= n || f0 >= fdim) return;
  int b = batch[node];
  float4 v = *(const float4*)&agg[(size_t)node * LD + f0];
  float* gb = g + (size_t)b * LD + f0;
  atomicAdd(gb + 0, v.x);
  if (f0 + 1 < fdim) atomicAdd(gb + 1, v.y);
  if (f0 + 2 < fdim) atomicAdd(gb + 2, v.z);
  if (f0 + 3 < fdim) atomicAdd(gb + 3, v.w);
}

// ---------------- tiny MLP head: one thread per graph ----------------
__global__ void k_mlp(const float* __restrict__ g,
                      const float* __restrict__ Wf1, const float* __restrict__ bf1,
                      const float* __restrict__ Wf2, const float* __restrict__ bf2,
                      const float* __restrict__ Wf3, const float* __restrict__ bf3,
                      const float* __restrict__ Wo,  const float* __restrict__ bo,
                      float* __restrict__ out, int ngraphs,
                      int din, int d1, int d2, int d3)
{
  int r = blockIdx.x * blockDim.x + threadIdx.x;
  if (r >= ngraphs) return;
  float t0[50], t1[30], t2[20], t3[10];
  for (int i = 0; i < din; ++i) t0[i] = g[(size_t)r * LD + i];
  for (int j = 0; j < d1; ++j) {
    float a = bf1[j];
    for (int i = 0; i < din; ++i) a += t0[i] * Wf1[i * d1 + j];
    t1[j] = a;
  }
  for (int j = 0; j < d2; ++j) {
    float a = bf2[j];
    for (int i = 0; i < d1; ++i) a += t1[i] * Wf2[i * d2 + j];
    t2[j] = a;
  }
  for (int j = 0; j < d3; ++j) {
    float a = bf3[j];
    for (int i = 0; i < d2; ++i) a += t2[i] * Wf3[i * d3 + j];
    t3[j] = a;
  }
  float a = bo[0];
  for (int i = 0; i < d3; ++i) a += t3[i] * Wo[i];
  out[r] = a;
}

extern "C" void kernel_launch(void* const* d_in, const int* in_sizes, int n_in,
                              void* d_out, int out_size, void* d_ws, size_t ws_size,
                              hipStream_t stream)
{
  (void)n_in; (void)ws_size;
  const float* x     = (const float*)d_in[0];
  const int*   ei    = (const int*)d_in[1];
  const int*   batch = (const int*)d_in[2];
  const float* W1  = (const float*)d_in[3];
  const float* b1  = (const float*)d_in[4];
  const float* W2  = (const float*)d_in[5];
  const float* b2  = (const float*)d_in[6];
  const float* Wf1 = (const float*)d_in[7];
  const float* bf1 = (const float*)d_in[8];
  const float* Wf2 = (const float*)d_in[9];
  const float* bf2 = (const float*)d_in[10];
  const float* Wf3 = (const float*)d_in[11];
  const float* bf3 = (const float*)d_in[12];
  const float* Wo  = (const float*)d_in[13];
  const float* bo  = (const float*)d_in[14];

  const int d1  = in_sizes[4];            // 60
  const int d2  = in_sizes[6];            // 50
  const int df1 = in_sizes[8];            // 30
  const int df2 = in_sizes[10];           // 20
  const int df3 = in_sizes[12];           // 10
  const int F   = in_sizes[3] / d1;       // 64
  const int N   = in_sizes[0] / F;        // 100000
  const long long E = in_sizes[1] / 2;    // 3.2M
  const int G   = out_size;               // 256

  const int* src = ei;
  const int* dst = ei + E;

  // workspace layout (two 25.6MB node buffers ping-pong)
  char* ws = (char*)d_ws;
  auto align512 = [](size_t v) { return (v + 511) & ~(size_t)511; };
  size_t o = 0;
  float* dinv = (float*)(ws + o); o = align512(o + (size_t)N * 4);
  float* gbuf = (float*)(ws + o); o = align512(o + (size_t)G * LD * 4);
  float* hbuf = (float*)(ws + o); o = align512(o + (size_t)N * LD * 4);
  float* abuf = (float*)(ws + o); o = align512(o + (size_t)N * LD * 4);

  const long long nt16 = (long long)N * 16;
  const long long et16 = E * 16;

  // ---- GCN normalization: deg = 1 + in-degree; dinv = rsqrt(deg)
  k_fill_ones<<<(N + 255) / 256, 256, 0, stream>>>(dinv, N);
  k_count_deg<<<(unsigned)((E + 255) / 256), 256, 0, stream>>>(dst, dinv, E);
  k_rsqrt_inplace<<<(N + 255) / 256, 256, 0, stream>>>(dinv, N);

  // ---- conv1: h1 = x @ W1 ; agg1 = h1*dinv^2 + b1 (fused) ; then edge scatter
  k_gemm_fused<<<(N + 63) / 64, 128, 0, stream>>>(x, F, F, W1, d1, dinv, b1, hbuf, abuf, N);
  k_edge_scatter<<<(unsigned)((et16 + 255) / 256), 256, 0, stream>>>(src, dst, dinv, hbuf, abuf, d1, E);

  // ---- conv2: h2 = agg1 @ W2 (into hbuf) ; agg2 init fused (into abuf, agg1 dead after read)
  k_gemm_fused<<<(N + 63) / 64, 128, 0, stream>>>(abuf, LD, d1, W2, d2, dinv, b2, hbuf, abuf, N);
  k_edge_scatter<<<(unsigned)((et16 + 255) / 256), 256, 0, stream>>>(src, dst, dinv, hbuf, abuf, d2, E);

  // ---- global_add_pool
  long long gelems = (long long)G * LD;
  k_zero<<<(unsigned)((gelems + 255) / 256), 256, 0, stream>>>(gbuf, gelems);
  k_pool<<<(unsigned)((nt16 + 255) / 256), 256, 0, stream>>>(batch, abuf, gbuf, d2, N);

  // ---- MLP head
  k_mlp<<<(G + 255) / 256, 256, 0, stream>>>(gbuf, Wf1, bf1, Wf2, bf2, Wf3, bf3,
                                             Wo, bo, (float*)d_out, G, d2, df1, df2, df3);
}
